// MPNNModel_69784628625437
// MI455X (gfx1250) — matrix operand order repaired and checked by reading, not compiled, not run on hardware
//
#include <hip/hip_runtime.h>

typedef __attribute__((ext_vector_type(16))) __bf16 v16bf;
typedef __attribute__((ext_vector_type(8)))  float  v8f;

#define DIMD 64
#define DIM2D 128

union Frag16 { v16bf v; uint4 q[2]; };

__device__ __forceinline__ unsigned short f2bfbits(float f) {
    unsigned int u = __builtin_bit_cast(unsigned int, f);
    return (unsigned short)((u + 0x7FFFu + ((u >> 16) & 1u)) >> 16);
}

// Pack two f32 -> two bf16 in one dword.
// v_perm_b32 selects the high 16 bits of each float after a +0x8000 rounding bias:
// 1 add + 1 add + 1 perm = 3 VALU per 2 elements.
#if __has_builtin(__builtin_amdgcn_perm)
__device__ __forceinline__ unsigned pack2bf(float a, float b) {
    unsigned ua = __builtin_bit_cast(unsigned, a) + 0x8000u;
    unsigned ub = __builtin_bit_cast(unsigned, b) + 0x8000u;
    // result bytes: {ub[3],ub[2],ua[3],ua[2]}  (s0 = ub -> pool bytes 4..7, s1 = ua -> 0..3)
    return __builtin_amdgcn_perm(ub, ua, 0x07060302u);
}
#else
__device__ __forceinline__ unsigned pack2bf(float a, float b) {
    return (unsigned)f2bfbits(a) | ((unsigned)f2bfbits(b) << 16);
}
#endif

__device__ __forceinline__ float fast_tanh(float x) {
#if __has_builtin(__builtin_amdgcn_tanhf)
    return __builtin_amdgcn_tanhf(x);       // single v_tanh_f32 (CDNA5 trans op)
#else
    // tanh(x) = 1 - 2/(e^(2x)+1);  e^(2x) = exp2(x * 2*log2(e))
    float e = __builtin_amdgcn_exp2f(x * 2.885390081777926815f);
    return 1.0f - 2.0f * __builtin_amdgcn_rcpf(e + 1.0f);
#endif
}

// Build one A fragment (16x32 bf16 slice) directly in registers.
// Lane (m,half): elements 0..7  = row cols base+0..7
//                elements 8..15 = row cols base+16..23   (base = kb + half*8)
__device__ __forceinline__ Frag16 make_a_frag(const float* __restrict__ base) {
    float4 lo0 = *(const float4*)(base);
    float4 lo1 = *(const float4*)(base + 4);
    float4 hi0 = *(const float4*)(base + 16);
    float4 hi1 = *(const float4*)(base + 20);
    Frag16 a;
    a.q[0].x = pack2bf(lo0.x, lo0.y);
    a.q[0].y = pack2bf(lo0.z, lo0.w);
    a.q[0].z = pack2bf(lo1.x, lo1.y);
    a.q[0].w = pack2bf(lo1.z, lo1.w);
    a.q[1].x = pack2bf(hi0.x, hi0.y);
    a.q[1].y = pack2bf(hi0.z, hi0.w);
    a.q[1].z = pack2bf(hi1.x, hi1.y);
    a.q[1].w = pack2bf(hi1.z, hi1.w);
    return a;
}

// Convert 4 weight matrices [128,64] f32 (row-major) -> bf16 transposed [64][128]
__global__ void convert_weights_kernel(const float* __restrict__ W0,
                                       const float* __restrict__ W1,
                                       const float* __restrict__ W2,
                                       const float* __restrict__ W3,
                                       unsigned short* __restrict__ out) {
    int i = blockIdx.x * blockDim.x + threadIdx.x;
    if (i >= 4 * DIM2D * DIMD) return;
    int w = i >> 13;
    int j = i & 8191;
    int k = j >> 6;            // 0..127
    int n = j & 63;            // 0..63
    const float* W = (w == 0) ? W0 : (w == 1) ? W1 : (w == 2) ? W2 : W3;
    out[w * 8192 + n * DIM2D + k] = f2bfbits(W[k * DIMD + n]);
}

// Message kernel: grid-stride over 16-edge tiles.
// msg = tanh([h[src]|ef] @ Wm + bm), atomically accumulated into pooled[dst].
__global__ __launch_bounds__(128) void msg_kernel(
    const float* __restrict__ h, const float* __restrict__ ef,
    const int* __restrict__ src, const int* __restrict__ dst,
    const unsigned short* __restrict__ WT,   // bf16 bits, [n][k] = [64][128]
    const float* __restrict__ bias,
    float* __restrict__ pooled, int nEdges)
{
    __shared__ unsigned short sW[DIMD * DIM2D];      // 16 KB

    const int tid  = threadIdx.x;
    const int lane = tid & 31;
    const int wave = tid >> 5;

    {   // cooperative weight load: 16 KB = 1024 uint4, 128 threads x 8
        const uint4* gw = (const uint4*)WT;
        uint4* lw = (uint4*)sW;
        #pragma unroll
        for (int i = 0; i < 8; ++i) lw[tid + 128 * i] = gw[tid + 128 * i];
    }
    __syncthreads();

    const int m    = lane & 15;
    const int half = lane >> 4;

    // bias for this lane's 4 output columns is tile-invariant: hoist
    float bn[4];
    #pragma unroll
    for (int t = 0; t < 4; ++t) bn[t] = bias[t * 16 + m];

    const long nTiles = nEdges / 16;
    const long wId    = (long)blockIdx.x * 4 + wave;
    const long nW     = (long)gridDim.x * 4;

    for (long tile = wId; tile < nTiles; tile += nW) {
        const long ebase = tile * 16;

        // Per-lane row: lane (m,half) owns A-matrix row m of this tile.
        const long eRow = ebase + m;
        const int  sRow = src[eRow];                     // gather index
        const float* hrow = h  + (long)sRow * DIMD;
        const float* erow = ef + eRow * DIMD;
        const int coff = half * 8;

        // Build all 4 A fragments straight from global memory (no LDS staging).
        Frag16 afr[4];
        afr[0] = make_a_frag(hrow + 0  + coff);   // concat cols   0..31
        afr[1] = make_a_frag(hrow + 32 + coff);   // concat cols  32..63
        afr[2] = make_a_frag(erow + 0  + coff);   // concat cols  64..95
        afr[3] = make_a_frag(erow + 32 + coff);   // concat cols 96..127

        v8f acc[4] = {};
        #pragma unroll
        for (int ks = 0; ks < 4; ++ks) {
            const int kb = ks * 32;
            #pragma unroll
            for (int t = 0; t < 4; ++t) {
                // B 32x16: lane (N=t*16+m) holds K kb+half*8 .. +7 and kb+16+half*8 .. +7
                Frag16 b;
                const unsigned short* bp = sW + (t * 16 + m) * DIM2D + kb + half * 8;
                b.q[0] = *(const uint4*)bp;
                b.q[1] = *(const uint4*)(bp + 16);
                acc[t] = __builtin_amdgcn_wmma_f32_16x16x32_bf16(
                    false, afr[ks].v, false, b.v, (short)0, acc[t], false, false);
            }
        }

        // C layout: acc[t][r] holds (M = r + half*8, N = t*16 + m)
        int dnode[8];
        #pragma unroll
        for (int r = 0; r < 8; ++r) dnode[r] = dst[ebase + r + half * 8];
        #pragma unroll
        for (int t = 0; t < 4; ++t) {
            const int n = t * 16 + m;
            #pragma unroll
            for (int r = 0; r < 8; ++r) {
                const float val = fast_tanh(acc[t][r] + bn[t]);
                __hip_atomic_fetch_add(&pooled[(long)dnode[r] * DIMD + n], val,
                                       __ATOMIC_RELAXED, __HIP_MEMORY_SCOPE_AGENT);
            }
        }
    }
}

// Update kernel: grid-stride over 16-node tiles. h_out = tanh([h_in|pooled] @ Wu + bu)
__global__ __launch_bounds__(128) void upd_kernel(
    const float* __restrict__ hin, const float* __restrict__ pooled,
    const unsigned short* __restrict__ WT, const float* __restrict__ bias,
    float* __restrict__ hout, int nNodes)
{
    __shared__ unsigned short sW[DIMD * DIM2D];

    const int tid  = threadIdx.x;
    const int lane = tid & 31;
    const int wave = tid >> 5;

    {
        const uint4* gw = (const uint4*)WT;
        uint4* lw = (uint4*)sW;
        #pragma unroll
        for (int i = 0; i < 8; ++i) lw[tid + 128 * i] = gw[tid + 128 * i];
    }
    __syncthreads();

    const int m    = lane & 15;
    const int half = lane >> 4;

    float bn[4];
    #pragma unroll
    for (int t = 0; t < 4; ++t) bn[t] = bias[t * 16 + m];

    const long nTiles = nNodes / 16;
    const long wId    = (long)blockIdx.x * 4 + wave;
    const long nW     = (long)gridDim.x * 4;

    for (long tile = wId; tile < nTiles; tile += nW) {
        const long nbase = tile * 16;
        const long node  = nbase + m;                    // this lane's A row
        const float* hrow = hin    + node * DIMD;
        const float* prow = pooled + node * DIMD;
        const int coff = half * 8;

        Frag16 afr[4];
        afr[0] = make_a_frag(hrow + 0  + coff);
        afr[1] = make_a_frag(hrow + 32 + coff);
        afr[2] = make_a_frag(prow + 0  + coff);
        afr[3] = make_a_frag(prow + 32 + coff);

        v8f acc[4] = {};
        #pragma unroll
        for (int ks = 0; ks < 4; ++ks) {
            const int kb = ks * 32;
            #pragma unroll
            for (int t = 0; t < 4; ++t) {
                Frag16 b;
                const unsigned short* bp = sW + (t * 16 + m) * DIM2D + kb + half * 8;
                b.q[0] = *(const uint4*)bp;
                b.q[1] = *(const uint4*)(bp + 16);
                acc[t] = __builtin_amdgcn_wmma_f32_16x16x32_bf16(
                    false, afr[ks].v, false, b.v, (short)0, acc[t], false, false);
            }
        }

        #pragma unroll
        for (int t = 0; t < 4; ++t) {
            const int n = t * 16 + m;
            #pragma unroll
            for (int r = 0; r < 8; ++r) {
                const int M = r + half * 8;
                hout[(nbase + M) * DIMD + n] = fast_tanh(acc[t][r] + bn[t]);
            }
        }
    }
}

// Final readout: out[i] = h2[i,:] . Wo + bo
__global__ void out_kernel(const float* __restrict__ h,
                           const float* __restrict__ Wo,
                           const float* __restrict__ bo,
                           float* __restrict__ out, int n) {
    int i = blockIdx.x * blockDim.x + threadIdx.x;
    if (i >= n) return;
    float s = bo[0];
    const float4* hp = (const float4*)(h + (long)i * DIMD);
    const float4* wp = (const float4*)Wo;
    #pragma unroll
    for (int j = 0; j < 16; ++j) {
        float4 a = hp[j], b = wp[j];
        s += a.x * b.x + a.y * b.y + a.z * b.z + a.w * b.w;
    }
    out[i] = s;
}

extern "C" void kernel_launch(void* const* d_in, const int* in_sizes, int n_in,
                              void* d_out, int out_size, void* d_ws, size_t ws_size,
                              hipStream_t stream) {
    const float* node_feats = (const float*)d_in[0];
    const float* edge_feats = (const float*)d_in[1];
    const int*   src        = (const int*)d_in[2];
    const int*   dst        = (const int*)d_in[3];
    const float* Wm1 = (const float*)d_in[4];
    const float* bm1 = (const float*)d_in[5];
    const float* Wu1 = (const float*)d_in[6];
    const float* bu1 = (const float*)d_in[7];
    const float* Wm2 = (const float*)d_in[8];
    const float* bm2 = (const float*)d_in[9];
    const float* Wu2 = (const float*)d_in[10];
    const float* bu2 = (const float*)d_in[11];
    const float* Wo  = (const float*)d_in[12];
    const float* bo  = (const float*)d_in[13];

    const int N = in_sizes[0] / DIMD;
    const int E = in_sizes[2];

    char* ws = (char*)d_ws;
    unsigned short* WT = (unsigned short*)ws;                 // 4 x 8192 ushort = 64 KB
    const size_t nodeBytes = (size_t)N * DIMD * sizeof(float);
    float* pooled = (float*)(ws + 65536);
    float* h1     = (float*)(ws + 65536 + nodeBytes);
    float* h2     = (float*)(ws + 65536 + 2 * nodeBytes);
    float* outp   = (float*)d_out;

    // Convert weights: slot 0 = Wm1, 1 = Wu1, 2 = Wm2, 3 = Wu2
    convert_weights_kernel<<<(4 * 8192 + 255) / 256, 256, 0, stream>>>(
        Wm1, Wu1, Wm2, Wu2, WT);

    // Device-filling persistent grids; waves iterate over tiles.
    const int tilesE  = E / 16;
    const int tilesN  = N / 16;
    int blocksE = (tilesE + 3) / 4; if (blocksE > 2048) blocksE = 2048;
    int blocksN = (tilesN + 3) / 4; if (blocksN > 512)  blocksN = 512;

    // Layer 1
    hipMemsetAsync(pooled, 0, nodeBytes, stream);
    msg_kernel<<<blocksE, 128, 0, stream>>>(node_feats, edge_feats, src, dst,
                                            WT + 0 * 8192, bm1, pooled, E);
    upd_kernel<<<blocksN, 128, 0, stream>>>(node_feats, pooled,
                                            WT + 1 * 8192, bu1, h1, N);
    // Layer 2
    hipMemsetAsync(pooled, 0, nodeBytes, stream);
    msg_kernel<<<blocksE, 128, 0, stream>>>(h1, edge_feats, src, dst,
                                            WT + 2 * 8192, bm2, pooled, E);
    upd_kernel<<<blocksN, 128, 0, stream>>>(h1, pooled,
                                            WT + 3 * 8192, bu2, h2, N);
    // Readout
    out_kernel<<<(N + 255) / 256, 256, 0, stream>>>(h2, Wo, bo, outp, N);
}
